// GNNLayer_57810259804276
// MI455X (gfx1250) — matrix operand order repaired and checked by reading, not compiled, not run on hardware
//
#include <hip/hip_runtime.h>
#include <hip/hip_bf16.h>
#include <math.h>

typedef __attribute__((ext_vector_type(16))) _Float16 v16h;
typedef __attribute__((ext_vector_type(8)))  float    v8f;

#define N_NODE 50000
#define N_EDGE 640000
#define DIM    128
#define REL_VOCAB (2*500 + 3)   // 1003
#define RT_PER_WAVE 4           // row tiles per wave in the GEMM

// ---------------------------------------------------------------------------
// Zero-fill (accumulator lives in poisoned workspace; re-zero every call).
// ---------------------------------------------------------------------------
__global__ void zero_kernel(float4* __restrict__ p, int n4) {
    int i = blockIdx.x * blockDim.x + threadIdx.x;
    if (i < n4) p[i] = make_float4(0.f, 0.f, 0.f, 0.f);
}

// ---------------------------------------------------------------------------
// One-shot: convert a 128x128 fp32 weight matrix into f16 stored in exact
// B-fragment order, so the GEMM hot loop loads each lane's 16 B values as a
// single contiguous 32-byte vector (2 x global_load_b128, zero cvt).
//
// B fragment layout (CDNA5 ISA 7.12.2, 16-bit B 32x16, wave32):
//   lane L: col = L%16, elems 0..7 = W[kt*32 + (L/16)*8 + i][col], elems 8..15 = +16
// Stored as Wf[(kt*32 + lane)*16 + i].
// ---------------------------------------------------------------------------
__global__ void convert_w_frag(const float* __restrict__ W,
                               _Float16* __restrict__ Wf) {
    const int t    = threadIdx.x;        // 256 threads, 1 block
    const int p    = t >> 1;             // 0..127 : (kt*32 + lane)
    const int h8   = t & 1;              // which 8-element half
    const int kt   = p >> 5;
    const int lane = p & 31;
    const int col  = lane & 15;
    const int kb   = kt * 32 + (lane >> 4) * 8 + h8 * 16;
    _Float16* dst  = Wf + p * 16 + h8 * 8;
    #pragma unroll
    for (int i = 0; i < 8; ++i)
        dst[i] = (_Float16)W[(kb + i) * DIM + col];
}

// ---------------------------------------------------------------------------
// C[M x 128] = A[M x 128] @ W[128 x 128]; A fp32 (cvt on the fly), W
// pre-converted f16 fragments, f32 accumulate. Each wave: 4 row tiles x 1
// column tile = 16 x v_wmma_f32_16x16x32_f16, B fragment reused across the
// 4 row tiles. 8 waves/block cover the 8 column tiles.
//
// A fragment (16x32 f16): lane L: row = L%16;
//   elems 0..7  = A[row][kt*32 + (L/16)*8 + i], elems 8..15 = +16.
// C fragment (16x16 f32): lane L: N = L%16; VGPR v: M = (L/16)*8 + v.
// ---------------------------------------------------------------------------
__global__ void gemm128_wmma(const float* __restrict__ A,
                             const _Float16* __restrict__ Wf,
                             float* __restrict__ C, int M) {
    const int lane = threadIdx.x & 31;
    const int wv   = threadIdx.x >> 5;   // 0..7 : column tile
    const int half = lane >> 4;          // 0 / 1
    const int l16  = lane & 15;
    const int rtg  = blockIdx.x;         // group of RT_PER_WAVE row tiles
    const int col  = wv * 16 + l16;

    v8f acc[RT_PER_WAVE] = {};

    #pragma unroll
    for (int kt = 0; kt < 4; ++kt) {
        const v16h b = *(const v16h*)(Wf + (kt * 32 + lane) * 16);
        #pragma unroll
        for (int rs = 0; rs < RT_PER_WAVE; ++rs) {
            int row = (rtg * RT_PER_WAVE + rs) * 16 + l16;
            int rA  = row < M ? row : (M - 1);   // clamp tail loads
            const float* ap = A + (long)rA * DIM + kt * 32 + half * 8;
            v16h a;
            #pragma unroll
            for (int i = 0; i < 8; ++i) {
                a[i]     = (_Float16)ap[i];
                a[i + 8] = (_Float16)ap[i + 16];
            }
            acc[rs] = __builtin_amdgcn_wmma_f32_16x16x32_f16(
                false, a, false, b, (short)0, acc[rs], false, false);
        }
    }

    #pragma unroll
    for (int rs = 0; rs < RT_PER_WAVE; ++rs) {
        const int m0 = (rtg * RT_PER_WAVE + rs) * 16 + half * 8;
        #pragma unroll
        for (int v = 0; v < 8; ++v) {
            int m = m0 + v;
            if (m < M) C[(long)m * DIM + col] = acc[rs][v];
        }
    }
}

// ---------------------------------------------------------------------------
// Edge kernel: one wave32 per edge. Lane k owns elements [4k, 4k+4).
//   alpha = sigmoid( sum_j relu(Ps[sub,j] + Pr[rel,j]) * w_alpha[j] + b )
//   atomicAdd(agg[obj, :], alpha * (hidden[sub,:] + rela[rel,:]))
// Indices scalarized via readfirstlane so gathers use SADDR-form b128 loads.
// ---------------------------------------------------------------------------
__global__ void edge_kernel(const float* __restrict__ Ps,
                            const float* __restrict__ Pr,
                            const float* __restrict__ hidden,
                            const float* __restrict__ rela,
                            const int*   __restrict__ edges,
                            const float* __restrict__ w_alpha,
                            const float* __restrict__ w_b,
                            float* __restrict__ agg, int n_edge) {
    const int lane = threadIdx.x & 31;
    const int wv   = threadIdx.x >> 5;
    const int e    = blockIdx.x * 8 + wv;
    if (e >= n_edge) return;

    const int rel = __builtin_amdgcn_readfirstlane(edges[e * 6 + 2]);
    const int sub = __builtin_amdgcn_readfirstlane(edges[e * 6 + 4]);
    const int obj = __builtin_amdgcn_readfirstlane(edges[e * 6 + 5]);
    const int k   = lane * 4;

    const float4 ps = *(const float4*)(Ps + (long)sub * DIM + k);
    const float4 pr = *(const float4*)(Pr + (long)rel * DIM + k);
    const float4 wa = *(const float4*)(w_alpha + k);

    float partial =
        fmaxf(ps.x + pr.x, 0.f) * wa.x +
        fmaxf(ps.y + pr.y, 0.f) * wa.y +
        fmaxf(ps.z + pr.z, 0.f) * wa.z +
        fmaxf(ps.w + pr.w, 0.f) * wa.w;

    #pragma unroll
    for (int m = 16; m >= 1; m >>= 1)
        partial += __shfl_xor(partial, m, 32);

    const float alpha = 1.0f / (1.0f + __expf(-(partial + w_b[0])));

    const float4 h = *(const float4*)(hidden + (long)sub * DIM + k);
    const float4 r = *(const float4*)(rela   + (long)rel * DIM + k);

    float* dst = agg + (long)obj * DIM + k;
    unsafeAtomicAdd(dst + 0, alpha * (h.x + r.x));
    unsafeAtomicAdd(dst + 1, alpha * (h.y + r.y));
    unsafeAtomicAdd(dst + 2, alpha * (h.z + r.z));
    unsafeAtomicAdd(dst + 3, alpha * (h.w + r.w));
}

// ---------------------------------------------------------------------------
extern "C" void kernel_launch(void* const* d_in, const int* in_sizes, int n_in,
                              void* d_out, int out_size, void* d_ws, size_t ws_size,
                              hipStream_t stream) {
    const float* hidden     = (const float*)d_in[0];
    const int*   edges      = (const int*)  d_in[1];
    // d_in[2] = n_node scalar, d_in[3] = old_nodes_new_idx (unused by output)
    const float* rela_embed = (const float*)d_in[4];
    const float* Ws_attn    = (const float*)d_in[5];
    const float* Wr_attn    = (const float*)d_in[6];
    const float* w_alpha_w  = (const float*)d_in[7];
    const float* w_alpha_b  = (const float*)d_in[8];
    const float* W_h        = (const float*)d_in[9];
    float*       out        = (float*)d_out;

    // Workspace layout (offsets all 64B-aligned)
    float*     agg  = (float*)d_ws;                         // 50000*128 f32
    float*     Ps   = agg + (size_t)N_NODE * DIM;           // 50000*128 f32
    float*     Pr   = Ps  + (size_t)N_NODE * DIM;           // 1008*128 f32
    _Float16*  WsF  = (_Float16*)(Pr + (size_t)1008 * DIM); // 16384 f16
    _Float16*  WrF  = WsF + 16384;
    _Float16*  WhF  = WrF + 16384;

    // 0) pre-convert weights into f16 fragment order (tiny one-off kernels)
    convert_w_frag<<<1, 256, 0, stream>>>(Ws_attn, WsF);
    convert_w_frag<<<1, 256, 0, stream>>>(Wr_attn, WrF);
    convert_w_frag<<<1, 256, 0, stream>>>(W_h,     WhF);

    // 1) zero accumulator
    {
        int n4 = N_NODE * DIM / 4;
        zero_kernel<<<(n4 + 255) / 256, 256, 0, stream>>>((float4*)agg, n4);
    }

    const int ROWS_PER_BLK = 16 * RT_PER_WAVE;  // 64
    // 2) Ps = hidden @ Ws_attn  (gather commutes with the GEMM)
    gemm128_wmma<<<(N_NODE + ROWS_PER_BLK - 1) / ROWS_PER_BLK, 256, 0, stream>>>(
        hidden, WsF, Ps, N_NODE);
    // 3) Pr = rela_embed @ Wr_attn
    gemm128_wmma<<<(REL_VOCAB + ROWS_PER_BLK - 1) / ROWS_PER_BLK, 256, 0, stream>>>(
        rela_embed, WrF, Pr, REL_VOCAB);
    // 4) per-edge attention + scatter-add
    edge_kernel<<<(N_EDGE + 7) / 8, 256, 0, stream>>>(Ps, Pr, hidden, rela_embed,
                                                      edges, w_alpha_w, w_alpha_b,
                                                      agg, N_EDGE);
    // 5) out = agg @ W_h
    gemm128_wmma<<<(N_NODE + ROWS_PER_BLK - 1) / ROWS_PER_BLK, 256, 0, stream>>>(
        agg, WhF, out, N_NODE);
}
